// RNNStateEncoder_61555471286996
// MI455X (gfx1250) — matrix operand order repaired and checked by reading, not compiled, not run on hardware
//
#include <hip/hip_runtime.h>

// ---------------------------------------------------------------------------
// Masked GRU (RNNStateEncoder) for MI455X / gfx1250, wave32 + WMMA f16->f32.
//   T=256 steps, N=128 batch, D=H=1024, gates G=3H=3072.
// Kernel 1: x_proj = x @ W_ih^T + b_ih        (big parallel WMMA GEMM)
// Kernel 2: seed h0 (f32->f16) + reset grid barrier
// Kernel 3: persistent GRU scan: 64 blocks keep their W_hh gate tiles in LDS
//           (96 KB of the 320 KB WGP LDS) for all 256 steps; h ping-pongs in
//           f16 through global; steps separated by an atomic grid barrier.
// ---------------------------------------------------------------------------

typedef __attribute__((ext_vector_type(16))) _Float16 v16h;
typedef __attribute__((ext_vector_type(8)))  _Float16 v8h;
typedef __attribute__((ext_vector_type(4)))  _Float16 v4h;
typedef __attribute__((ext_vector_type(8)))  float    v8f;

#define GRU_T 256
#define GRU_N 128
#define GRU_D 1024
#define GRU_H 1024
#define GRU_G 3072   // 3*H
#define SCAN_BLOCKS (GRU_H / 16)   // 64 persistent blocks

// Convert a float4 to 4 halves and store 8B (ds_store_b64 / LDS).
__device__ __forceinline__ void st4h(_Float16* p, float4 f) {
  v4h h;
  h[0] = (_Float16)f.x; h[1] = (_Float16)f.y;
  h[2] = (_Float16)f.z; h[3] = (_Float16)f.w;
  *(v4h*)p = h;
}

// Load a 16-half WMMA fragment as two 16B reads.
__device__ __forceinline__ v16h ld_frag(const _Float16* base, int off_lo, int off_hi) {
  v8h lo = *(const v8h*)(base + off_lo);
  v8h hi = *(const v8h*)(base + off_hi);
  return __builtin_shufflevector(lo, hi, 0,1,2,3,4,5,6,7,8,9,10,11,12,13,14,15);
}

__device__ __forceinline__ float fsig(float x) {
  return __builtin_amdgcn_rcpf(1.0f + __expf(-x));   // v_exp_f32 + v_rcp_f32
}
__device__ __forceinline__ float ftanh(float x) {
  return 2.0f * fsig(2.0f * x) - 1.0f;
}

// ---------------------------------------------------------------------------
// Kernel 1: C[M,G] = X[M,K] * W[G,K]^T + bias[G]
// 256 threads (8 waves), block tile 128x128, K staged in 32s.
// grid = (G/128, M/128). Per wave per K-chunk: 1 A frag, 8 B frags, 8 WMMAs.
// ---------------------------------------------------------------------------
__global__ __launch_bounds__(256)
void xproj_gemm(const float* __restrict__ X, const float* __restrict__ W,
                const float* __restrict__ bias, float* __restrict__ C) {
  __shared__ _Float16 Alds[128 * 32];  // 8 KB
  __shared__ _Float16 Blds[128 * 32];  // 8 KB

  const int tid  = threadIdx.x;
  const int wave = tid >> 5;
  const int lane = tid & 31;
  const int m16  = lane & 15;
  const int half = lane >> 4;
  const int row0 = blockIdx.y * 128;
  const int col0 = blockIdx.x * 128;

  v8f acc[8] = {};

  for (int k0 = 0; k0 < GRU_D; k0 += 32) {
#pragma unroll
    for (int i = 0; i < 4; ++i) {           // stage A: 128x32 f32 -> f16
      int idx = i * 256 + tid;
      int r = idx >> 3, q = idx & 7;
      float4 f = *(const float4*)(X + (size_t)(row0 + r) * GRU_D + k0 + q * 4);
      st4h(&Alds[r * 32 + q * 4], f);
    }
#pragma unroll
    for (int i = 0; i < 4; ++i) {           // stage B: 128x32 f32 -> f16
      int idx = i * 256 + tid;
      int r = idx >> 3, q = idx & 7;
      float4 f = *(const float4*)(W + (size_t)(col0 + r) * GRU_D + k0 + q * 4);
      st4h(&Blds[r * 32 + q * 4], f);
    }
    __syncthreads();

    if (k0 + 32 < GRU_D) {
      __builtin_prefetch(X + (size_t)(row0 + (tid >> 1)) * GRU_D + k0 + 32, 0, 0);
    }

    v16h a = ld_frag(&Alds[(wave * 16 + m16) * 32], 8 * half, 16 + 8 * half);
#pragma unroll
    for (int ct = 0; ct < 8; ++ct) {
      v16h b = ld_frag(&Blds[(ct * 16 + m16) * 32], 8 * half, 16 + 8 * half);
      acc[ct] = __builtin_amdgcn_wmma_f32_16x16x32_f16(
          false, a, false, b, (short)0, acc[ct], false, false);
    }
    __syncthreads();
  }

#pragma unroll
  for (int ct = 0; ct < 8; ++ct) {
    int gcol = col0 + ct * 16 + m16;
    float bv = bias[gcol];
#pragma unroll
    for (int r = 0; r < 8; ++r) {
      int grow = row0 + wave * 16 + half * 8 + r;
      C[(size_t)grow * GRU_G + gcol] = acc[ct][r] + bv;
    }
  }
}

// ---------------------------------------------------------------------------
// Kernel 2: seed h0 -> f16 ping buffer; reset grid-barrier counter.
// ---------------------------------------------------------------------------
__global__ __launch_bounds__(256)
void seed_kernel(const float* __restrict__ h0, _Float16* __restrict__ h16,
                 unsigned* __restrict__ bar) {
  int i = blockIdx.x * 256 + threadIdx.x;    // grid covers N*H
  h16[i] = (_Float16)h0[i];
  if (i == 0) *bar = 0u;
}

// ---------------------------------------------------------------------------
// Monotonic grid barrier (64 co-resident blocks).
// ---------------------------------------------------------------------------
__device__ __forceinline__ void grid_barrier(unsigned* bar, unsigned target) {
  __threadfence();
  __syncthreads();
  if (threadIdx.x == 0) {
    __hip_atomic_fetch_add(bar, 1u, __ATOMIC_ACQ_REL, __HIP_MEMORY_SCOPE_AGENT);
    while (__hip_atomic_load(bar, __ATOMIC_ACQUIRE, __HIP_MEMORY_SCOPE_AGENT) < target) {
      __builtin_amdgcn_s_sleep(2);
    }
  }
  __syncthreads();
}

// ---------------------------------------------------------------------------
// Kernel 3: persistent GRU scan.
// Block b owns h-columns [b*16, b*16+16). Prologue stages its 3 gate weight
// tiles (48 x 1024, f16) into LDS once. Each step: per wave, K-loop of
// {2 global b128 A-frag loads from f16 h, 6 ds_load_b128 B frags, 3 WMMAs},
// then f32 gate math, writes out + f16 h ping-pong, grid barrier.
// Dynamic LDS: 3*16*1024 halves = 96 KB.
// ---------------------------------------------------------------------------
__global__ __launch_bounds__(256)
void gru_scan(_Float16* __restrict__ h16a, _Float16* __restrict__ h16b,
              const float* __restrict__ xproj, const float* __restrict__ Whh,
              const float* __restrict__ bhh, const int* __restrict__ masks,
              float* __restrict__ out, unsigned* __restrict__ bar) {
  extern __shared__ _Float16 Bt[];           // [48][1024] f16 = 96 KB

  const int tid  = threadIdx.x;
  const int wave = tid >> 5;
  const int lane = tid & 31;
  const int m16  = lane & 15;
  const int half = lane >> 4;
  const int col0 = blockIdx.x * 16;

  // ---- prologue: stage W_hh rows {j, H+j, 2H+j} for j in col tile (f32->f16)
  for (int idx = tid; idx < 48 * (GRU_H / 4); idx += 256) {   // 12288 float4
    int r = idx >> 8, q = idx & 255;          // r in [0,48), q in [0,256)
    int gate = r >> 4, j = r & 15;
    float4 f = *(const float4*)(Whh + (size_t)(gate * GRU_H + col0 + j) * GRU_H + q * 4);
    st4h(&Bt[r * GRU_H + q * 4], f);
  }
  __syncthreads();

  const int j   = col0 + m16;                 // this lane's output column
  const float br_ = bhh[j];
  const float bz_ = bhh[GRU_H + j];
  const float bn_ = bhh[2 * GRU_H + j];
  const int arow = wave * 16 + m16;           // A-fragment row for this lane
  const v16h zfrag = {};

  for (int t = 0; t < GRU_T; ++t) {
    const _Float16* hin  = (t & 1) ? h16b : h16a;
    _Float16*       hout = (t & 1) ? h16a : h16b;
    const int*   mask = masks + (size_t)t * GRU_N;
    const float* xpt  = xproj + (size_t)t * GRU_N * GRU_G;
    float*       outt = out + (size_t)t * GRU_N * GRU_H;

    v8f ar = {}, az = {}, an = {};
    const bool mkeep = (mask[arow] != 0);
    const _Float16* hrow = hin + (size_t)arow * GRU_H;

#pragma unroll 4
    for (int k0 = 0; k0 < GRU_H; k0 += 32) {
      v16h a = ld_frag(hrow, k0 + 16 * half, k0 + 16 * half + 8);
      a = mkeep ? a : zfrag;                  // masked hidden state
      v16h b0 = ld_frag(&Bt[(0 * 16 + m16) * GRU_H], k0 + 16 * half, k0 + 16 * half + 8);
      v16h b1 = ld_frag(&Bt[(1 * 16 + m16) * GRU_H], k0 + 16 * half, k0 + 16 * half + 8);
      v16h b2 = ld_frag(&Bt[(2 * 16 + m16) * GRU_H], k0 + 16 * half, k0 + 16 * half + 8);
      ar = __builtin_amdgcn_wmma_f32_16x16x32_f16(false, a, false, b0, (short)0, ar, false, false);
      az = __builtin_amdgcn_wmma_f32_16x16x32_f16(false, a, false, b1, (short)0, az, false, false);
      an = __builtin_amdgcn_wmma_f32_16x16x32_f16(false, a, false, b2, (short)0, an, false, false);
    }

    // ---- gate math: lane holds rows n = wave*16 + 8*half + r, column j
#pragma unroll
    for (int r = 0; r < 8; ++r) {
      int n = wave * 16 + half * 8 + r;
      const float* xpr = xpt + (size_t)n * GRU_G;
      float xr = xpr[j];
      float xz = xpr[GRU_H + j];
      float xn = xpr[2 * GRU_H + j];
      float mf = (mask[n] != 0) ? 1.0f : 0.0f;
      float hp = (float)hin[(size_t)n * GRU_H + j] * mf;

      float rg = fsig(xr + ar[r] + br_);
      float zg = fsig(xz + az[r] + bz_);
      float nc = ftanh(xn + rg * (an[r] + bn_));
      float hn = (1.0f - zg) * nc + zg * hp;

      outt[(size_t)n * GRU_H + j] = hn;
      hout[(size_t)n * GRU_H + j] = (_Float16)hn;
      if (t == GRU_T - 1) {
        out[(size_t)GRU_T * GRU_N * GRU_H + (size_t)n * GRU_H + j] = hn;  // h_last
      }
    }

    if (t != GRU_T - 1) {
      grid_barrier(bar, (unsigned)SCAN_BLOCKS * (unsigned)(t + 1));
    }
  }
}

// ---------------------------------------------------------------------------
extern "C" void kernel_launch(void* const* d_in, const int* in_sizes, int n_in,
                              void* d_out, int out_size, void* d_ws, size_t ws_size,
                              hipStream_t stream) {
  const float* x     = (const float*)d_in[0];   // (T*N, D)
  const float* h0    = (const float*)d_in[1];   // (N, 1, H)
  const int*   masks = (const int*)  d_in[2];   // (T*N,)
  const float* W_ih  = (const float*)d_in[3];   // (3H, D)
  const float* W_hh  = (const float*)d_in[4];   // (3H, H)
  const float* b_ih  = (const float*)d_in[5];   // (3H,)
  const float* b_hh  = (const float*)d_in[6];   // (3H,)
  float* out = (float*)d_out;                   // (T*N,H) then (N,1,H)

  // workspace: x_proj f32 | h16 ping | h16 pong | barrier counter
  float*    xproj = (float*)d_ws;
  _Float16* h16a  = (_Float16*)(xproj + (size_t)GRU_T * GRU_N * GRU_G);
  _Float16* h16b  = h16a + (size_t)GRU_N * GRU_H;
  unsigned* bar   = (unsigned*)(h16b + (size_t)GRU_N * GRU_H);

  // x_proj = x @ W_ih^T + b_ih
  dim3 g1(GRU_G / 128, (GRU_T * GRU_N) / 128);
  xproj_gemm<<<g1, 256, 0, stream>>>(x, W_ih, b_ih, xproj);

  // seed h0 as f16, reset barrier
  seed_kernel<<<(GRU_N * GRU_H) / 256, 256, 0, stream>>>(h0, h16a, bar);

  // persistent scan (96 KB dynamic LDS per block)
  gru_scan<<<SCAN_BLOCKS, 256, 3 * 16 * GRU_H * sizeof(_Float16), stream>>>(
      h16a, h16b, xproj, W_hh, b_hh, masks, out, bar);
}
